// FIBERCrossModalFusion_11544872091794
// MI455X (gfx1250) — matrix operand order repaired and checked by reading, not compiled, not run on hardware
//
#include <hip/hip_runtime.h>
#include <math.h>

// ---------------------------------------------------------------------------
// FIBER cross-modal fusion for MI455X (gfx1250), bf16 WMMA GEMM core with
// GLOBAL_LOAD_ASYNC_TO_LDS_B128 double-buffered staging (ASYNCcnt tracked).
// ---------------------------------------------------------------------------

typedef __attribute__((ext_vector_type(16))) __bf16 v16bf;
typedef __attribute__((ext_vector_type(8)))  float  v8f;

#define Dm   1024
#define Sm   512
#define Bm   32
#define Hm   16
#define FFm  4096
#define Lm   3

// ---------------- deterministic abs-mean scale ----------------
__global__ __launch_bounds__(256) void absmean_part(const float* __restrict__ w,
                                                    int n, float* __restrict__ part) {
  __shared__ float red[256];
  float s = 0.f;
  for (int i = blockIdx.x * 256 + threadIdx.x; i < n; i += gridDim.x * 256)
    s += fabsf(w[i]);
  red[threadIdx.x] = s; __syncthreads();
  for (int st = 128; st > 0; st >>= 1) {
    if (threadIdx.x < st) red[threadIdx.x] += red[threadIdx.x + st];
    __syncthreads();
  }
  if (threadIdx.x == 0) part[blockIdx.x] = red[0];
}

__global__ __launch_bounds__(256) void absmean_final(const float* __restrict__ part,
                                                     int nb, float inv_n,
                                                     float* __restrict__ scale) {
  __shared__ float red[256];
  float s = 0.f;
  for (int i = threadIdx.x; i < nb; i += 256) s += part[i];
  red[threadIdx.x] = s; __syncthreads();
  for (int st = 128; st > 0; st >>= 1) {
    if (threadIdx.x < st) red[threadIdx.x] += red[threadIdx.x + st];
    __syncthreads();
  }
  if (threadIdx.x == 0) {
    float sc = red[0] * inv_n;
    sc = fminf(fmaxf(sc, 1e-5f), 1000.f);
    scale[0] = sc;
  }
}

// ternary quantize: q in {-1,0,+1} stored exactly as bf16
__global__ __launch_bounds__(256) void quantize_kernel(const float* __restrict__ w,
                                                       const float* __restrict__ scale_p,
                                                       __bf16* __restrict__ q, int n) {
  float s = scale_p[0];
  const float thr = 2.f / 3.f;
  for (int i = blockIdx.x * 256 + threadIdx.x; i < n; i += gridDim.x * 256) {
    float wn = w[i] / s;           // clip(+-10) cannot change the comparisons
    float qv = (wn > thr) ? 1.f : ((wn < -thr) ? -1.f : 0.f);
    q[i] = (__bf16)qv;
  }
}

// ---------------- f32 -> bf16 ----------------
__global__ __launch_bounds__(256) void f32_to_bf16(const float* __restrict__ x,
                                                   __bf16* __restrict__ y, int n) {
  int i = blockIdx.x * 256 + threadIdx.x;
  if (i < n) y[i] = (__bf16)x[i];
}

// v2 = kv2[:, 1024:2048] -> bf16
__global__ __launch_bounds__(256) void halfslice_bf16(const float* __restrict__ kv2,
                                                      __bf16* __restrict__ v2b, int n) {
  int i = blockIdx.x * 256 + threadIdx.x;
  if (i < n) {
    int b = i >> 10, j = i & 1023;
    v2b[i] = (__bf16)kv2[b * 2048 + 1024 + j];
  }
}

// ---------------- WMMA bf16 GEMM: C[M,N] = A[M,K] x W[N,K]^T -----------------
// EPI: 0 = f32 (scale*acc+bias); 1 = bf16; 2 = exact-gelu -> bf16;
//      3 = f32 residual add; 4 = residual add -> f32 AND bf16 dual-store
#define LSTR 40   // padded LDS row stride in bf16 elems (80B, 16B aligned)

// issue per-lane async global->LDS copies for one 128x32 bf16 tile pair
__device__ __forceinline__ void issue_async_tile(const __bf16* __restrict__ A,
                                                 const __bf16* __restrict__ W,
                                                 long M, long K, long rowBase,
                                                 long colBase, long kk, int tid,
                                                 unsigned la_base, unsigned lb_base) {
  const int lrow = tid >> 1;
  const int lk   = (tid & 1) << 4;                 // 0 or 16 bf16 elems
  const unsigned loff = (unsigned)(lrow * LSTR + lk) * 2u;  // byte offset in tile
  const unsigned la = la_base + loff;
  const unsigned lb = lb_base + loff;
  // W tile: always in range (N multiple of 128, K multiple of 32)
  {
    const __bf16* g = W + (colBase + lrow) * K + kk + lk;
    asm volatile(
        "global_load_async_to_lds_b128 %0, %2, off\n\t"
        "global_load_async_to_lds_b128 %1, %3, off"
        :: "v"(lb), "v"(lb + 16u),
           "v"((unsigned long long)(uintptr_t)g),
           "v"((unsigned long long)(uintptr_t)(g + 8))
        : "memory");
  }
  // A tile: predicate off out-of-range rows (their LDS rows are pre-zeroed)
  if (rowBase + lrow < M) {
    const __bf16* g = A + (rowBase + lrow) * K + kk + lk;
    asm volatile(
        "global_load_async_to_lds_b128 %0, %2, off\n\t"
        "global_load_async_to_lds_b128 %1, %3, off"
        :: "v"(la), "v"(la + 16u),
           "v"((unsigned long long)(uintptr_t)g),
           "v"((unsigned long long)(uintptr_t)(g + 8))
        : "memory");
  }
}

__device__ __forceinline__ void wait_async_all() {
  asm volatile("s_wait_asynccnt 0x0" ::: "memory");
}

__global__ __launch_bounds__(256, 1)
void gemm_bitnet_wmma(const __bf16* __restrict__ A,   // [M,K] row-major bf16
                      const __bf16* __restrict__ W,   // [N,K] row-major ternary bf16
                      const float* __restrict__ scale_p,
                      const float* __restrict__ bias, // [N]
                      const float* __restrict__ res,  // [M,N] for EPI 3/4
                      float* __restrict__ outF,
                      __bf16* __restrict__ outB,
                      int M, int N, int K, int epi) {
  __shared__ __bf16 lsA[2][128 * LSTR];
  __shared__ __bf16 lsB[2][128 * LSTR];

  const int tid  = threadIdx.x;
  const int lane = tid & 31;
  const int wave = tid >> 5;
  const int row0 = (wave >> 2) * 64;     // wave M origin within block tile
  const int col0 = (wave & 3) * 32;      // wave N origin within block tile
  const long rowBase = (long)blockIdx.y * 128;
  const long colBase = (long)blockIdx.x * 128;
  const long Kl = K, Ml = M;

  const unsigned laddrA[2] = { (unsigned)(uintptr_t)&lsA[0][0],
                               (unsigned)(uintptr_t)&lsA[1][0] };
  const unsigned laddrB[2] = { (unsigned)(uintptr_t)&lsB[0][0],
                               (unsigned)(uintptr_t)&lsB[1][0] };

  // pre-zero LDS rows of A that have no source row (async copies are predicated off)
  if (rowBase + 128 > Ml) {
    const int lrow = tid >> 1;
    const int lk   = (tid & 1) << 4;
    if (rowBase + lrow >= Ml) {
      uint4 z = make_uint4(0u, 0u, 0u, 0u);
#pragma unroll
      for (int b = 0; b < 2; ++b) {
        *(uint4*)(&lsA[b][0] + lrow * LSTR + lk)     = z;
        *(uint4*)(&lsA[b][0] + lrow * LSTR + lk + 8) = z;
      }
    }
  }

  v8f acc[4][2] = {};

  // prologue: async-stage k=0 into buffer 0
  issue_async_tile(A, W, Ml, Kl, rowBase, colBase, 0, tid, laddrA[0], laddrB[0]);
  wait_async_all();
  __syncthreads();

  const int nk = K >> 5;
  int buf = 0;
  for (int ks = 0; ks < nk; ++ks) {
    if (ks + 1 < nk)   // fire-and-forget DMA into the back buffer
      issue_async_tile(A, W, Ml, Kl, rowBase, colBase, (long)(ks + 1) << 5, tid,
                       laddrA[buf ^ 1], laddrB[buf ^ 1]);

    const __bf16* As = lsA[buf];
    const __bf16* Bs = lsB[buf];

    // A fragments: lanes0-15 row m, elems K0-15;
    //              lanes16-31 row m, elems K16-31
    const int koffA = (lane < 16) ? 0 : 16;
    const int koffB = (lane < 16) ? 0 : 16;  // B: lanes0-15 K0-15, lanes16-31 K16-31
    union { uint4 u[2]; v16bf v; } fa[4], fb[2];
#pragma unroll
    for (int n = 0; n < 2; ++n) {
      int c = col0 + n * 16 + (lane & 15);
      fb[n].u[0] = *(const uint4*)(Bs + c * LSTR + koffB);
      fb[n].u[1] = *(const uint4*)(Bs + c * LSTR + koffB + 8);
    }
#pragma unroll
    for (int m = 0; m < 4; ++m) {
      int r = row0 + m * 16 + (lane & 15);
      fa[m].u[0] = *(const uint4*)(As + r * LSTR + koffA);
      fa[m].u[1] = *(const uint4*)(As + r * LSTR + koffA + 8);
    }
#pragma unroll
    for (int m = 0; m < 4; ++m)
#pragma unroll
      for (int n = 0; n < 2; ++n)
        acc[m][n] = __builtin_amdgcn_wmma_f32_16x16x32_bf16(
            false, fa[m].v, false, fb[n].v, (short)0, acc[m][n], false, false);

    wait_async_all();   // own wave's DMA into back buffer done
    __syncthreads();    // publish to all waves
    buf ^= 1;
  }

  // epilogue
  const float sc = scale_p[0];
#pragma unroll
  for (int m = 0; m < 4; ++m) {
#pragma unroll
    for (int n = 0; n < 2; ++n) {
      long rb0 = rowBase + row0 + m * 16 + ((lane < 16) ? 0 : 8);
      long c   = colBase + col0 + n * 16 + (lane & 15);
      float bcol = bias[c];
#pragma unroll
      for (int j = 0; j < 8; ++j) {
        long r = rb0 + j;
        if (r < Ml) {
          float val = acc[m][n][j] * sc + bcol;
          long off = r * (long)N + c;
          if (epi == 0) {
            outF[off] = val;
          } else if (epi == 1) {
            outB[off] = (__bf16)val;
          } else if (epi == 2) {
            float g = 0.5f * val * (1.f + erff(val * 0.70710678118654752f));
            outB[off] = (__bf16)g;
          } else if (epi == 3) {
            outF[off] = res[off] + val;
          } else {  // epi == 4: residual, dual f32 + bf16 store
            float t = res[off] + val;
            outF[off] = t;
            outB[off] = (__bf16)t;
          }
        }
      }
    }
  }
}

// ---------------- LayerNorm (D=1024, 256 threads/row, two-pass) ----------------
// resMode: 0 none, 1 full residual res[row], 2 broadcast residual res[row/Sdim]
__global__ __launch_bounds__(256)
void layernorm_kernel(const float* __restrict__ x, const float* __restrict__ res,
                      int resMode, int Sdim,
                      const float* __restrict__ g, const float* __restrict__ bta,
                      float* __restrict__ outF, __bf16* __restrict__ outB) {
  __shared__ float red[256];
  const long row = blockIdx.x;
  const int tid = threadIdx.x;
  const float* xr = x + row * Dm;
  const float* rr = nullptr;
  if (resMode == 1)      rr = res + row * Dm;
  else if (resMode == 2) rr = res + (row / Sdim) * Dm;

  float v[4]; float s = 0.f;
#pragma unroll
  for (int i = 0; i < 4; ++i) {
    int j = tid + i * 256;
    float t = xr[j];
    if (rr) t += rr[j];
    v[i] = t; s += t;
  }
  red[tid] = s; __syncthreads();
  for (int st = 128; st > 0; st >>= 1) { if (tid < st) red[tid] += red[tid + st]; __syncthreads(); }
  float mean = red[0] * (1.f / 1024.f);
  __syncthreads();
  float q = 0.f;
#pragma unroll
  for (int i = 0; i < 4; ++i) { float d = v[i] - mean; q += d * d; }
  red[tid] = q; __syncthreads();
  for (int st = 128; st > 0; st >>= 1) { if (tid < st) red[tid] += red[tid + st]; __syncthreads(); }
  float rstd = rsqrtf(red[0] * (1.f / 1024.f) + 1e-5f);
#pragma unroll
  for (int i = 0; i < 4; ++i) {
    int j = tid + i * 256;
    float o = (v[i] - mean) * rstd * g[j] + bta[j];
    if (outF) outF[row * Dm + j] = o;
    if (outB) outB[row * Dm + j] = (__bf16)o;
  }
}

// ---------------- image->text attention (one query per image) ----------------
// grid = B*H blocks; kv is bf16 [B,S,2D]; out f32 [B,D]
__global__ __launch_bounds__(256)
void attention_kernel(const float* __restrict__ qf, const __bf16* __restrict__ kv,
                      const unsigned char* __restrict__ mask, float* __restrict__ out) {
  __shared__ float qs[64];
  __shared__ float lg[512];
  __shared__ float red[256];
  const int bh = blockIdx.x, b = bh >> 4, h = bh & 15;
  const int tid = threadIdx.x;
  if (tid < 64) qs[tid] = qf[(long)b * Dm + h * 64 + tid];
  __syncthreads();
  for (int s = tid; s < Sm; s += 256) {
    const __bf16* kp = kv + (long)(b * Sm + s) * (2 * Dm) + h * 64;
    float a = 0.f;
#pragma unroll 8
    for (int d = 0; d < 64; ++d) a += qs[d] * (float)kp[d];
    a *= 0.125f;                                   // HD^-0.5
    if (!mask[b * Sm + s]) a = -INFINITY;
    lg[s] = a;
  }
  __syncthreads();
  red[tid] = fmaxf(lg[tid], lg[tid + 256]); __syncthreads();
  for (int st = 128; st > 0; st >>= 1) { if (tid < st) red[tid] = fmaxf(red[tid], red[tid + st]); __syncthreads(); }
  float mx = red[0];
  __syncthreads();
  float ps = 0.f;
  for (int s = tid; s < Sm; s += 256) { float e = expf(lg[s] - mx); lg[s] = e; ps += e; }
  red[tid] = ps; __syncthreads();
  for (int st = 128; st > 0; st >>= 1) { if (tid < st) red[tid] += red[tid + st]; __syncthreads(); }
  float inv = 1.f / red[0];
  __syncthreads();
  if (tid < 64) {
    float a = 0.f;
    const __bf16* vp = kv + (long)b * Sm * (2 * Dm) + Dm + h * 64 + tid;
    for (int s = 0; s < Sm; ++s) a += lg[s] * (float)vp[(long)s * (2 * Dm)];
    out[(long)b * Dm + h * 64 + tid] = a * inv;
  }
}

// ---------------------------------------------------------------------------
extern "C" void kernel_launch(void* const* d_in, const int* in_sizes, int n_in,
                              void* d_out, int out_size, void* d_ws, size_t ws_size,
                              hipStream_t stream) {
  (void)in_sizes; (void)n_in; (void)out_size;
  const size_t MB = 1ull << 20;
  if (ws_size < 362 * MB) return;  // required scratch (see layout below)

  const float* vision = (const float*)d_in[0];
  const float* text   = (const float*)d_in[1];
  const unsigned char* mask = (const unsigned char*)d_in[2];  // jnp bool = 1 byte
  // params flattened in dict-insertion order, 32 leaves per layer
  enum { VP_W = 0, VP_B, TP_W, TP_B, I2TQ_W, I2TQ_B, I2TKV_W, I2TKV_B,
         T2IQ_W, T2IQ_B, T2IKV_W, T2IKV_B, OUT_W, OUT_B,
         VN_G, VN_B, TN_G, TN_B, CN_G, CN_B, VFLN_G, VFLN_B, TFLN_G, TFLN_B,
         VF_W1, VF_B1, VF_W2, VF_B2, TF_W1, TF_B1, TF_W2, TF_B2 };
  auto pin = [&](int l, int k) -> const float* {
    return (const float*)d_in[3 + 32 * l + k];
  };

  // ---- workspace layout ----
  char* Wp = (char*)d_ws;
  float*  t_x   = (float*)(Wp + 0);            // 64MB  [16384,1024] f32 residual stream
  float*  buf1  = (float*)(Wp + 64 * MB);      // 64MB  f32 GEMM outs [16384,1024]
  __bf16* kvbf  = (__bf16*)(Wp + 128 * MB);    // 64MB  i2tkv bf16 [16384,2048]
  __bf16* actbf = (__bf16*)(Wp + 192 * MB);    // 32MB  bf16 A staging [16384,1024]
  __bf16* hbf   = (__bf16*)(Wp + 224 * MB);    // 128MB gelu bf16 [16384,4096]
  __bf16* qw    = (__bf16*)(Wp + 352 * MB);    // 8MB   quantized weight (max 4096x1024)
  char* sm = Wp + 360 * MB;
  float*  v_x    = (float*)sm;  sm += 131072;
  __bf16* vbf    = (__bf16*)sm; sm += 65536;
  float*  vpf    = (float*)sm;  sm += 131072;
  __bf16* vplnbf = (__bf16*)sm; sm += 65536;
  float*  qf     = (float*)sm;  sm += 131072;
  float*  i2t    = (float*)sm;  sm += 131072;
  __bf16* i2tbf  = (__bf16*)sm; sm += 65536;
  float*  kv2    = (float*)sm;  sm += 262144;
  __bf16* v2bf   = (__bf16*)sm; sm += 65536;
  float*  i2tO   = (float*)sm;  sm += 131072;
  float*  t2iO   = (float*)sm;  sm += 131072;
  __bf16* vlnbf  = (__bf16*)sm; sm += 65536;
  __bf16* hvbf   = (__bf16*)sm; sm += 262144;
  float*  part   = (float*)sm;  sm += 4096;
  float*  scl    = (float*)sm;  sm += 256;

  const int TROWS = Bm * Sm;  // 16384

  auto quant = [&](const float* w, int n) {
    absmean_part<<<1024, 256, 0, stream>>>(w, n, part);
    absmean_final<<<1, 256, 0, stream>>>(part, 1024, 1.f / (float)n, scl);
    quantize_kernel<<<2048, 256, 0, stream>>>(w, scl, qw, n);
  };
  auto gemm = [&](const __bf16* A, const float* bias, const float* resid,
                  float* oF, __bf16* oB, int M, int N, int K, int epi) {
    dim3 grid(N / 128, (M + 127) / 128);
    gemm_bitnet_wmma<<<grid, 256, 0, stream>>>(A, qw, scl, bias, resid, oF, oB, M, N, K, epi);
  };
  auto to_bf = [&](const float* x, __bf16* y, int n) {
    f32_to_bf16<<<(n + 255) / 256, 256, 0, stream>>>(x, y, n);
  };

  // residual streams
  hipMemcpyAsync(v_x, vision, (size_t)Bm * Dm * 4, hipMemcpyDeviceToDevice, stream);
  hipMemcpyAsync(t_x, text, (size_t)TROWS * Dm * 4, hipMemcpyDeviceToDevice, stream);

  for (int l = 0; l < Lm; ++l) {
    // bf16 copies of the residual streams: layer 0 converts explicitly;
    // later layers get them fused from the previous FFN-2 epilogue (EPI 4).
    if (l == 0) {
      to_bf(v_x, vbf, Bm * Dm);
      to_bf(t_x, actbf, TROWS * Dm);
    }
    // vp = LN(bitnet(v)) -> bf16
    quant(pin(l, VP_W), Dm * Dm);
    gemm(vbf, pin(l, VP_B), nullptr, vpf, nullptr, Bm, Dm, Dm, 0);
    layernorm_kernel<<<Bm, 256, 0, stream>>>(vpf, nullptr, 0, Sm,
        pin(l, VN_G), pin(l, VN_B), nullptr, vplnbf);
    // tp = LN(bitnet(t)) -> bf16
    quant(pin(l, TP_W), Dm * Dm);
    gemm(actbf, pin(l, TP_B), nullptr, buf1, nullptr, TROWS, Dm, Dm, 0);
    layernorm_kernel<<<TROWS, 256, 0, stream>>>(buf1, nullptr, 0, Sm,
        pin(l, TN_G), pin(l, TN_B), nullptr, actbf);
    // kv = bitnet(tp) -> bf16 [16384,2048]
    quant(pin(l, I2TKV_W), 2 * Dm * Dm);
    gemm(actbf, pin(l, I2TKV_B), nullptr, nullptr, kvbf, TROWS, 2 * Dm, Dm, 1);
    // q = bitnet(vp)
    quant(pin(l, I2TQ_W), Dm * Dm);
    gemm(vplnbf, pin(l, I2TQ_B), nullptr, qf, nullptr, Bm, Dm, Dm, 0);
    // attention -> i2t [32,1024]
    attention_kernel<<<Bm * Hm, 256, 0, stream>>>(qf, kvbf, mask, i2t);
    to_bf(i2t, i2tbf, Bm * Dm);
    // kv2 = bitnet(vp) [32,2048]; v2 = second half
    quant(pin(l, T2IKV_W), 2 * Dm * Dm);
    gemm(vplnbf, pin(l, T2IKV_B), nullptr, kv2, nullptr, Bm, 2 * Dm, Dm, 0);
    halfslice_bf16<<<(Bm * Dm + 255) / 256, 256, 0, stream>>>(kv2, v2bf, Bm * Dm);
    // out projections: i2tO = out(i2t); t2iO = out(v2) (broadcast shortcut)
    quant(pin(l, OUT_W), Dm * Dm);
    gemm(i2tbf, pin(l, OUT_B), nullptr, i2tO, nullptr, Bm, Dm, Dm, 0);
    gemm(v2bf,  pin(l, OUT_B), nullptr, t2iO, nullptr, Bm, Dm, Dm, 0);
    // v = LN(i2tO + v);  t = LN(broadcast(t2iO) + t)
    layernorm_kernel<<<Bm, 256, 0, stream>>>(i2tO, v_x, 1, Sm,
        pin(l, CN_G), pin(l, CN_B), v_x, nullptr);
    layernorm_kernel<<<TROWS, 256, 0, stream>>>(t_x, t2iO, 2, Sm,
        pin(l, CN_G), pin(l, CN_B), t_x, nullptr);
    // vision FFN
    layernorm_kernel<<<Bm, 256, 0, stream>>>(v_x, nullptr, 0, Sm,
        pin(l, VFLN_G), pin(l, VFLN_B), nullptr, vlnbf);
    quant(pin(l, VF_W1), FFm * Dm);
    gemm(vlnbf, pin(l, VF_B1), nullptr, nullptr, hvbf, Bm, FFm, Dm, 2);
    quant(pin(l, VF_W2), Dm * FFm);
    gemm(hvbf, pin(l, VF_B2), v_x, v_x, vbf, Bm, Dm, FFm, 4);      // + bf16 v for next layer
    // text FFN
    layernorm_kernel<<<TROWS, 256, 0, stream>>>(t_x, nullptr, 0, Sm,
        pin(l, TFLN_G), pin(l, TFLN_B), nullptr, actbf);
    quant(pin(l, TF_W1), FFm * Dm);
    gemm(actbf, pin(l, TF_B1), nullptr, nullptr, hbf, TROWS, FFm, Dm, 2);
    quant(pin(l, TF_W2), Dm * FFm);
    gemm(hbf, pin(l, TF_B2), t_x, t_x, actbf, TROWS, Dm, FFm, 4);  // + bf16 t for next layer
  }

  // outputs: v [32,1024] then t [32,512,1024], f32, concatenated
  hipMemcpyAsync(d_out, v_x, (size_t)Bm * Dm * 4, hipMemcpyDeviceToDevice, stream);
  hipMemcpyAsync((char*)d_out + (size_t)Bm * Dm * 4, t_x, (size_t)TROWS * Dm * 4,
                 hipMemcpyDeviceToDevice, stream);
}
